// MultiLatentAttention_11158325035223
// MI455X (gfx1250) — compile-verified, matrix operand order checked by reading, not run on hardware
//
#include <hip/hip_runtime.h>
#include <hip/hip_bf16.h>
#include <stdint.h>

// ---------------- problem constants ----------------
#define BATCH   2
#define SEQ     2048
#define EMBED   2048
#define HEADS   16
#define NOPE    128
#define VDIM    128
#define ROPED   64
#define Q_HEAD  192          // NOPE + ROPE
#define KV_RANK 512
#define BH      (BATCH * HEADS)
#define TOKENS  (BATCH * SEQ)   // 4096

typedef __attribute__((ext_vector_type(16))) __bf16 v16bf;
typedef __attribute__((ext_vector_type(8)))  float  v8f;

// ---------------- small helpers ----------------
__device__ __forceinline__ unsigned short f2bf(float f) {
    unsigned int u = __float_as_uint(f);
    u += 0x7FFFu + ((u >> 16) & 1u);        // round-to-nearest-even
    return (unsigned short)(u >> 16);
}

__device__ __forceinline__ v8f v8f_zero() {
    v8f z = {0.f, 0.f, 0.f, 0.f, 0.f, 0.f, 0.f, 0.f};
    return z;
}

// Load one 16x32 bf16 fragment (A- or B-operand) from row-major storage
// [rows][ld] (bf16 as u16). Follows the CDNA5 16-bit operand layout:
// lane L -> row r0 + (L&15); element e -> K = (e/8)*16 + (L>>4)*8 + (e%8).
// => two contiguous 16B loads per lane (global_load_b128 or ds_load_b128).
__device__ __forceinline__ v16bf load_frag16x32(const unsigned short* base,
                                                int ld, int r0, int k0) {
    const int lane = threadIdx.x & 31;
    const int row  = r0 + (lane & 15);
    const int half = lane >> 4;
    const unsigned short* p = base + (size_t)row * (size_t)ld + k0 + half * 8;
    union { struct { uint4 lo; uint4 hi; } q; v16bf v; } u;
    u.q.lo = *(const uint4*)(p);
    u.q.hi = *(const uint4*)(p + 16);
    return u.v;
}

__device__ __forceinline__ v8f wmma_bf16(v16bf a, v16bf b, v8f c) {
    // D = A(16x32) * B(32x16) + C, f32 accum
    return __builtin_amdgcn_wmma_f32_16x16x32_bf16(
        /*neg_a=*/false, a, /*neg_b=*/false, b,
        /*c_mod=*/(short)0, c, /*reuse_a=*/false, /*reuse_b=*/false);
}

// Async-copy one 64x32 bf16 slab (64 rows x 64B) from global [rows][ldg] into
// an LDS tile [64][32], using GLOBAL_LOAD_ASYNC_TO_LDS_B128 (ASYNCcnt-tracked).
// Each instruction moves 32 lanes x 16B = 8 contiguous rows; 8 instructions/slab.
// LDS byte address = low 32 bits of the generic shared-pointer (aperture map).
__device__ __forceinline__ void async_copy_slab(const unsigned short* __restrict__ g,
                                                int ldg, int row0, int k0,
                                                unsigned short* lds_tile) {
    const int lane  = threadIdx.x & 31;
    const int rsub  = lane >> 2;   // 0..7
    const int chunk = lane & 3;    // 0..3, 16B each
#pragma unroll
    for (int i = 0; i < 8; ++i) {
        const int row = i * 8 + rsub;
        const unsigned short* gp =
            g + (size_t)(row0 + row) * (size_t)ldg + k0 + chunk * 8;
        unsigned lds_addr =
            (unsigned)(uintptr_t)(lds_tile + row * 32 + chunk * 8);
        asm volatile("global_load_async_to_lds_b128 %0, %1, off"
                     :
                     : "v"(lds_addr), "v"((unsigned long long)(uintptr_t)gp)
                     : "memory");
    }
}

// ---------------- conversion / layout kernels ----------------
__global__ void cvt_f32_bf16(const float* __restrict__ src,
                             unsigned short* __restrict__ dst, size_t n) {
    size_t i = (size_t)blockIdx.x * blockDim.x + threadIdx.x;
    if (i < n) dst[i] = f2bf(src[i]);
}

// W [K,N] f32 -> Wt [N,K] bf16
__global__ void transpose_cvt(const float* __restrict__ W,
                              unsigned short* __restrict__ Wt,
                              int K, int N) {
    size_t i = (size_t)blockIdx.x * blockDim.x + threadIdx.x;
    size_t total = (size_t)K * (size_t)N;
    if (i >= total) return;
    int k = (int)(i / N);
    int n = (int)(i % N);
    Wt[(size_t)n * K + k] = f2bf(W[i]);
}

// ckv [TOKENS,576] f32 -> compressed bf16 [TOKENS,512] + roped k_pe bf16 [TOKENS,64]
__global__ void prep_ckv(const float* __restrict__ ckv,
                         unsigned short* __restrict__ comp,
                         unsigned short* __restrict__ krope) {
    const int token = blockIdx.x;
    const int pos   = token % SEQ;
    const float* src = ckv + (size_t)token * (KV_RANK + ROPED);
    for (int i = threadIdx.x; i < KV_RANK; i += 64)
        comp[(size_t)token * KV_RANK + i] = f2bf(src[i]);
    if (threadIdx.x < ROPED / 2) {
        int i = threadIdx.x;
        float theta = __powf(10000.0f, -2.0f * (float)i / (float)ROPED);
        float sn, cs;
        __sincosf((float)pos * theta, &sn, &cs);
        float x1 = src[KV_RANK + 2 * i];
        float x2 = src[KV_RANK + 2 * i + 1];
        krope[(size_t)token * ROPED + 2 * i]     = f2bf(x1 * cs - x2 * sn);
        krope[(size_t)token * ROPED + 2 * i + 1] = f2bf(x1 * sn + x2 * cs);
    }
}

// q [TOKENS, HEADS*192] f32 -> qf bf16 [BH][SEQ][192], rope last 64, * 1/sqrt(192)
__global__ void prep_q(const float* __restrict__ q,
                       unsigned short* __restrict__ qf) {
    const int token = blockIdx.x;
    const int h     = blockIdx.y;
    const int pos   = token % SEQ;
    const int b     = token / SEQ;
    const float scale = 0.07216878364870323f;   // 1/sqrt(192)
    const float* src = q + (size_t)token * (HEADS * Q_HEAD) + h * Q_HEAD;
    unsigned short* dst =
        qf + ((size_t)(b * HEADS + h) * SEQ + pos) * Q_HEAD;
    for (int i = threadIdx.x; i < NOPE; i += 64)
        dst[i] = f2bf(src[i] * scale);
    if (threadIdx.x < ROPED / 2) {
        int i = threadIdx.x;
        float theta = __powf(10000.0f, -2.0f * (float)i / (float)ROPED);
        float sn, cs;
        __sincosf((float)pos * theta, &sn, &cs);
        float x1 = src[NOPE + 2 * i];
        float x2 = src[NOPE + 2 * i + 1];
        dst[NOPE + 2 * i]     = f2bf((x1 * cs - x2 * sn) * scale);
        dst[NOPE + 2 * i + 1] = f2bf((x1 * sn + x2 * cs) * scale);
    }
}

// kv [TOKENS, HEADS*256] f32 + krope -> kf bf16 [BH][SEQ][192], vT bf16 [BH][128][SEQ]
__global__ void prep_kv(const float* __restrict__ kv,
                        const unsigned short* __restrict__ krope,
                        unsigned short* __restrict__ kf,
                        unsigned short* __restrict__ vT) {
    const int token = blockIdx.x;
    const int h     = blockIdx.y;
    const int pos   = token % SEQ;
    const int b     = token / SEQ;
    const int bh    = b * HEADS + h;
    const int d     = threadIdx.x;              // 0..127
    const float* src = kv + (size_t)token * (HEADS * (NOPE + VDIM)) + h * (NOPE + VDIM);
    unsigned short* kdst = kf + ((size_t)bh * SEQ + pos) * Q_HEAD;
    kdst[d] = f2bf(src[d]);                                      // k_nope
    if (d < ROPED) kdst[NOPE + d] = krope[(size_t)token * ROPED + d];
    vT[((size_t)bh * VDIM + d) * SEQ + pos] = f2bf(src[NOPE + d]);
}

// attention O f32 [BH][SEQ][128] -> ao bf16 [TOKENS][HEADS*128]
__global__ void prep_out(const float* __restrict__ O,
                         unsigned short* __restrict__ ao) {
    const int token = blockIdx.x;
    const int pos   = token % SEQ;
    const int b     = token / SEQ;
    for (int i = threadIdx.x; i < HEADS * VDIM; i += 256) {
        int h = i / VDIM, d = i % VDIM;
        ao[(size_t)token * (HEADS * VDIM) + i] =
            f2bf(O[((size_t)(b * HEADS + h) * SEQ + pos) * VDIM + d]);
    }
}

// ---------------- WMMA GEMM: C[M,N] = A[M,K] * Bt[N,K]^T ----------------
// one wave per 64x64 output tile; K stepped by 32.
// A/B k-slabs are staged into LDS double-buffers with async global->LDS
// copies (ASYNCcnt): issue slab k+1, s_wait_asynccnt 16 (previous slab
// landed), then ds_load fragments + 16 WMMAs overlap the in-flight copies.
template <int OUT_BF16>
__global__ void __launch_bounds__(32)
__attribute__((amdgpu_waves_per_eu(3, 8)))
gemm64x64(const unsigned short* __restrict__ A,
          const unsigned short* __restrict__ Bt,
          void* __restrict__ Cv,
          int M, int N, int K) {
    const int n0   = blockIdx.x * 64;
    const int m0   = blockIdx.y * 64;
    const int lane = threadIdx.x & 31;
    const int half = lane >> 4;
    const int col  = lane & 15;

    __shared__ __align__(16) unsigned short tA[2][64 * 32];
    __shared__ __align__(16) unsigned short tB[2][64 * 32];

    v8f acc[4][4];
#pragma unroll
    for (int i = 0; i < 4; ++i)
#pragma unroll
        for (int j = 0; j < 4; ++j) acc[i][j] = v8f_zero();

    const int nk = K >> 5;
    // prologue: stage slab 0
    async_copy_slab(A,  K, m0, 0, tA[0]);
    async_copy_slab(Bt, K, n0, 0, tB[0]);

    for (int ks = 0; ks < nk; ++ks) {
        const int buf = ks & 1;
        if (ks + 1 < nk) {
            async_copy_slab(A,  K, m0, (ks + 1) * 32, tA[buf ^ 1]);
            async_copy_slab(Bt, K, n0, (ks + 1) * 32, tB[buf ^ 1]);
            // <=16 outstanding: the 16 copies of slab ks have completed
            asm volatile("s_wait_asynccnt 16" ::: "memory");
        } else {
            asm volatile("s_wait_asynccnt 0" ::: "memory");
        }
        v16bf af[4];
#pragma unroll
        for (int i = 0; i < 4; ++i)
            af[i] = load_frag16x32(tA[buf], 32, i * 16, 0);
#pragma unroll
        for (int j = 0; j < 4; ++j) {
            v16bf bf = load_frag16x32(tB[buf], 32, j * 16, 0);
#pragma unroll
            for (int i = 0; i < 4; ++i)
                acc[i][j] = wmma_bf16(af[i], bf, acc[i][j]);
        }
    }

#pragma unroll
    for (int i = 0; i < 4; ++i)
#pragma unroll
        for (int j = 0; j < 4; ++j)
#pragma unroll
            for (int r = 0; r < 8; ++r) {
                int row = m0 + i * 16 + half * 8 + r;
                int c   = n0 + j * 16 + col;
                float v = acc[i][j][r];
                if (OUT_BF16)
                    ((unsigned short*)Cv)[(size_t)row * N + c] = f2bf(v);
                else
                    ((float*)Cv)[(size_t)row * N + c] = v;
            }
}

// ---------------- flash attention (causal) ----------------
// one wave handles 32 query rows of one (b,h); online softmax, P transposed
// via LDS. Q stays register-resident (K-fragments reused by both row-tiles).
// waves_per_eu(2,8): budget ~512 VGPRs for ~330 live -> no spills.
__global__ void __launch_bounds__(32)
__attribute__((amdgpu_waves_per_eu(2, 8)))
mla_attn(const unsigned short* __restrict__ qf,
         const unsigned short* __restrict__ kf,
         const unsigned short* __restrict__ vT,
         float* __restrict__ O) {
    const int bh   = blockIdx.y;
    const int q0   = blockIdx.x * 32;
    const int lane = threadIdx.x & 31;
    const int half = lane >> 4;
    const int col  = lane & 15;

    const unsigned short* qh = qf + (size_t)bh * SEQ * Q_HEAD;
    const unsigned short* kh = kf + (size_t)bh * SEQ * Q_HEAD;
    const unsigned short* vh = vT + (size_t)bh * VDIM * SEQ;

    __shared__ __align__(16) unsigned short pls[32][32];

    // keep Q fragments resident for the whole key loop (2 row-tiles x 6 K-steps)
    v16bf qa[2][6];
#pragma unroll
    for (int rt = 0; rt < 2; ++rt)
#pragma unroll
        for (int ks = 0; ks < 6; ++ks)
            qa[rt][ks] = load_frag16x32(qh, Q_HEAD, q0 + rt * 16, ks * 32);

    v8f o[2][8];
    float mrow[2][8], lrow[2][8];
#pragma unroll
    for (int rt = 0; rt < 2; ++rt) {
#pragma unroll
        for (int dt = 0; dt < 8; ++dt) o[rt][dt] = v8f_zero();
#pragma unroll
        for (int r = 0; r < 8; ++r) { mrow[rt][r] = -3.0e38f; lrow[rt][r] = 0.f; }
    }

    for (int kb = 0; kb <= q0; kb += 32) {
        // ---- S = Q * K^T for a 32x32 block (4 tiles x 6 K-steps) ----
        v8f s[2][2];
        s[0][0] = v8f_zero(); s[0][1] = v8f_zero();
        s[1][0] = v8f_zero(); s[1][1] = v8f_zero();
#pragma unroll
        for (int ct = 0; ct < 2; ++ct)
#pragma unroll
            for (int ks = 0; ks < 6; ++ks) {
                v16bf kb_frag = load_frag16x32(kh, Q_HEAD, kb + ct * 16, ks * 32);
                s[0][ct] = wmma_bf16(qa[0][ks], kb_frag, s[0][ct]);
                s[1][ct] = wmma_bf16(qa[1][ks], kb_frag, s[1][ct]);
            }

        // ---- causal mask + online softmax, P -> LDS (bf16) ----
#pragma unroll
        for (int rt = 0; rt < 2; ++rt) {
#pragma unroll
            for (int r = 0; r < 8; ++r) {
                const int row = q0 + rt * 16 + half * 8 + r;
                float v0 = s[rt][0][r];
                float v1 = s[rt][1][r];
                if (kb + col > row)      v0 = -3.0e38f;
                if (kb + 16 + col > row) v1 = -3.0e38f;
                float mx = fmaxf(v0, v1);
#pragma unroll
                for (int off = 1; off < 16; off <<= 1)
                    mx = fmaxf(mx, __shfl_xor(mx, off, 32));
                const float nm = fmaxf(mrow[rt][r], mx);
                const float fs = __expf(mrow[rt][r] - nm);
                mrow[rt][r] = nm;
                lrow[rt][r] *= fs;
#pragma unroll
                for (int dt = 0; dt < 8; ++dt) o[rt][dt][r] *= fs;
                const float p0 = __expf(v0 - nm);
                const float p1 = __expf(v1 - nm);
                float ps = p0 + p1;
#pragma unroll
                for (int off = 1; off < 16; off <<= 1)
                    ps += __shfl_xor(ps, off, 32);
                lrow[rt][r] += ps;
                pls[rt * 16 + half * 8 + r][col]      = f2bf(p0);
                pls[rt * 16 + half * 8 + r][16 + col] = f2bf(p1);
            }
        }
        __syncthreads();

        // ---- O += P(16x32) * V(32x16 per d-tile) ----
        v16bf pa0 = load_frag16x32(&pls[0][0], 32, 0, 0);
        v16bf pa1 = load_frag16x32(&pls[0][0], 32, 16, 0);
#pragma unroll
        for (int dt = 0; dt < 8; ++dt) {
            v16bf vb = load_frag16x32(vh, SEQ, dt * 16, kb);
            o[0][dt] = wmma_bf16(pa0, vb, o[0][dt]);
            o[1][dt] = wmma_bf16(pa1, vb, o[1][dt]);
        }
        __syncthreads();
    }

    // ---- normalize and store ----
#pragma unroll
    for (int rt = 0; rt < 2; ++rt)
#pragma unroll
        for (int r = 0; r < 8; ++r) {
            const float inv = 1.0f / lrow[rt][r];
            const int row = q0 + rt * 16 + half * 8 + r;
#pragma unroll
            for (int dt = 0; dt < 8; ++dt)
                O[((size_t)bh * SEQ + row) * VDIM + dt * 16 + col] =
                    o[rt][dt][r] * inv;
        }
}

// ---------------- host launcher ----------------
extern "C" void kernel_launch(void* const* d_in, const int* in_sizes, int n_in,
                              void* d_out, int out_size, void* d_ws, size_t ws_size,
                              hipStream_t stream) {
    const float* x    = (const float*)d_in[0];
    const float* Wqa  = (const float*)d_in[1];
    const float* Wqb  = (const float*)d_in[2];
    const float* Wkva = (const float*)d_in[3];
    const float* Wkvb = (const float*)d_in[4];
    const float* Wout = (const float*)d_in[5];

    // bump allocator over workspace
    char* ws = (char*)d_ws;
    size_t off = 0;
    auto alloc = [&](size_t bytes) -> void* {
        void* p = ws + off;
        off += (bytes + 255) & ~(size_t)255;
        return p;
    };
    unsigned short* xb     = (unsigned short*)alloc((size_t)TOKENS * EMBED * 2);
    unsigned short* WqaT   = (unsigned short*)alloc((size_t)512 * EMBED * 2);
    unsigned short* WqbT   = (unsigned short*)alloc((size_t)(HEADS * Q_HEAD) * 512 * 2);
    unsigned short* WkvaT  = (unsigned short*)alloc((size_t)(KV_RANK + ROPED) * EMBED * 2);
    unsigned short* WkvbT  = (unsigned short*)alloc((size_t)(HEADS * (NOPE + VDIM)) * KV_RANK * 2);
    unsigned short* WoutT  = (unsigned short*)alloc((size_t)EMBED * (HEADS * VDIM) * 2);
    unsigned short* cq_b   = (unsigned short*)alloc((size_t)TOKENS * 512 * 2);
    float*          q_f    = (float*)alloc((size_t)TOKENS * (HEADS * Q_HEAD) * 4);
    float*          ckv_f  = (float*)alloc((size_t)TOKENS * (KV_RANK + ROPED) * 4);
    unsigned short* comp_b = (unsigned short*)alloc((size_t)TOKENS * KV_RANK * 2);
    unsigned short* krop_b = (unsigned short*)alloc((size_t)TOKENS * ROPED * 2);
    float*          kv_f   = (float*)alloc((size_t)TOKENS * (HEADS * (NOPE + VDIM)) * 4);
    unsigned short* qf_b   = (unsigned short*)alloc((size_t)BH * SEQ * Q_HEAD * 2);
    unsigned short* kf_b   = (unsigned short*)alloc((size_t)BH * SEQ * Q_HEAD * 2);
    unsigned short* vT_b   = (unsigned short*)alloc((size_t)BH * VDIM * SEQ * 2);
    float*          O_f    = (float*)alloc((size_t)BH * SEQ * VDIM * 4);
    unsigned short* ao_b   = (unsigned short*)alloc((size_t)TOKENS * (HEADS * VDIM) * 2);
    (void)ws_size; (void)in_sizes; (void)n_in; (void)out_size;

    const int T256 = 256;
    auto blocks = [](size_t n, int t) { return (unsigned)((n + t - 1) / t); };

    // 1. convert inputs to bf16 (weights transposed to [N,K])
    cvt_f32_bf16<<<blocks((size_t)TOKENS * EMBED, T256), T256, 0, stream>>>(
        x, xb, (size_t)TOKENS * EMBED);
    transpose_cvt<<<blocks((size_t)EMBED * 512, T256), T256, 0, stream>>>(Wqa, WqaT, EMBED, 512);
    transpose_cvt<<<blocks((size_t)512 * (HEADS * Q_HEAD), T256), T256, 0, stream>>>(
        Wqb, WqbT, 512, HEADS * Q_HEAD);
    transpose_cvt<<<blocks((size_t)EMBED * (KV_RANK + ROPED), T256), T256, 0, stream>>>(
        Wkva, WkvaT, EMBED, KV_RANK + ROPED);
    transpose_cvt<<<blocks((size_t)KV_RANK * (HEADS * (NOPE + VDIM)), T256), T256, 0, stream>>>(
        Wkvb, WkvbT, KV_RANK, HEADS * (NOPE + VDIM));
    transpose_cvt<<<blocks((size_t)(HEADS * VDIM) * EMBED, T256), T256, 0, stream>>>(
        Wout, WoutT, HEADS * VDIM, EMBED);

    // 2. cq = x @ Wqa   (bf16 out: feeds next GEMM)
    gemm64x64<1><<<dim3(512 / 64, TOKENS / 64), 32, 0, stream>>>(
        xb, WqaT, cq_b, TOKENS, 512, EMBED);
    // 3. q = cq @ Wqb
    gemm64x64<0><<<dim3((HEADS * Q_HEAD) / 64, TOKENS / 64), 32, 0, stream>>>(
        cq_b, WqbT, q_f, TOKENS, HEADS * Q_HEAD, 512);
    // 4. ckv = x @ Wkva
    gemm64x64<0><<<dim3((KV_RANK + ROPED) / 64, TOKENS / 64), 32, 0, stream>>>(
        xb, WkvaT, ckv_f, TOKENS, KV_RANK + ROPED, EMBED);
    // 5. split ckv -> compressed bf16 + roped k_pe
    prep_ckv<<<TOKENS, 64, 0, stream>>>(ckv_f, comp_b, krop_b);
    // 6. kv = compressed @ Wkvb
    gemm64x64<0><<<dim3((HEADS * (NOPE + VDIM)) / 64, TOKENS / 64), 32, 0, stream>>>(
        comp_b, WkvbT, kv_f, TOKENS, HEADS * (NOPE + VDIM), KV_RANK);
    // 7. per-head layouts + RoPE + score scale folded into q
    prep_q<<<dim3(TOKENS, HEADS), 64, 0, stream>>>(q_f, qf_b);
    prep_kv<<<dim3(TOKENS, HEADS), 128, 0, stream>>>(kv_f, krop_b, kf_b, vT_b);
    // 8. causal flash attention
    mla_attn<<<dim3(SEQ / 32, BH), 32, 0, stream>>>(qf_b, kf_b, vT_b, O_f);
    // 9. reorder to [tokens, H*128] bf16
    prep_out<<<TOKENS, 256, 0, stream>>>(O_f, ao_b);
    // 10. final projection -> d_out (f32)
    gemm64x64<0><<<dim3(EMBED / 64, TOKENS / 64), 32, 0, stream>>>(
        ao_b, WoutT, (float*)d_out, TOKENS, EMBED, HEADS * VDIM);
}
